// CstrFlashModel_53077205844643
// MI455X (gfx1250) — matrix operand order repaired and checked by reading, not compiled, not run on hardware
//
#include <hip/hip_runtime.h>
#include <hip/hip_bf16.h>

typedef _Float16 v16h __attribute__((ext_vector_type(16)));
typedef _Float16 v8h  __attribute__((ext_vector_type(8)));
typedef float    v8f  __attribute__((ext_vector_type(8)));

#define RK_DELTA 0.01f

// ---------- fast, branch-free transcendentals on native HW ops ----------
// tanh(x) = 1 - 2/(exp(2x)+1); exp(2x) = exp2(x * 2*log2(e))
__device__ __forceinline__ float fast_tanh(float x) {
  float e = __builtin_amdgcn_exp2f(x * 2.8853900817779268f); // v_exp_f32
  float r = __builtin_amdgcn_rcpf(e + 1.0f);                 // v_rcp_f32
  return __builtin_fmaf(-2.0f, r, 1.0f);
}

// ---------- WMMA fragment loads from LDS ----------
// A-matrix 16x32 f16, contiguous k order used here (B matches): lane l: m=l&15, hi=l>>4,
//   halfs[0..15] = A[m][kblk + hi*16 + 0..15]
__device__ __forceinline__ v16h frag_a(const _Float16* X, int row, int kblk, int hi) {
  union { v16h v; v8h h[2]; } u;
  const _Float16* p = X + row * 64 + kblk + hi * 16;
  u.h[0] = *(const v8h*)p;
  u.h[1] = *(const v8h*)(p + 8);
  return u.v;
}
// B-matrix 32x16 f16 (contiguous k order, same as frag_a): lane l: n=l&15,
//   halfs[0..15] = B[kblk + hi*16 + 0..15][n]  -> contiguous when W stored as Wt[n][k]
__device__ __forceinline__ v16h frag_b(const _Float16* Wt, int n, int kblk, int hi) {
  return *(const v16h*)(Wt + n * 64 + kblk + hi * 16);
}

// tanh( X(32x64) @ Wt^T(64x64) + bias ) -> OUT(32x64) f16. One wave, 2 M-tiles x 4 N-tiles.
__device__ __forceinline__ void layer64(const _Float16* X, const _Float16* Wt,
                                        const float* bias, _Float16* OUT, int lane) {
  const int m  = lane & 15;
  const int hi = lane >> 4;
  for (int mt = 0; mt < 2; ++mt) {
    const int rbase = mt * 16;
    v16h a0 = frag_a(X, rbase + m, 0,  hi);
    v16h a1 = frag_a(X, rbase + m, 32, hi);
    for (int nt = 0; nt < 4; ++nt) {
      v16h b0 = frag_b(Wt, nt * 16 + m, 0,  hi);
      v16h b1 = frag_b(Wt, nt * 16 + m, 32, hi);
      const float bb = bias[nt * 16 + m];
      // bias folded into accumulator: C layout -> every vgpr of this lane is column n
      v8f c = {bb, bb, bb, bb, bb, bb, bb, bb};
      c = __builtin_amdgcn_wmma_f32_16x16x32_f16(false, a0, false, b0, (short)0, c, false, false);
      c = __builtin_amdgcn_wmma_f32_16x16x32_f16(false, a1, false, b1, (short)0, c, false, false);
      // D layout: element (m = v + 8*hi, n = lane&15)
      _Float16* orow = OUT + (rbase + 8 * hi) * 64 + nt * 16 + m;
#pragma unroll
      for (int v = 0; v < 8; ++v)
        orow[v * 64] = (_Float16)fast_tanh(c[v]);
    }
  }
}

// X(32x64) @ W3t^T(64x16, rows 8..15 zero) + b3 -> sF(32x8) f32
__device__ __forceinline__ void layer_out(const _Float16* X, const _Float16* W3t,
                                          const float* b3, float* sF, int lane) {
  const int m  = lane & 15;
  const int hi = lane >> 4;
  v16h b0 = frag_b(W3t, m, 0,  hi);
  v16h b1 = frag_b(W3t, m, 32, hi);
  const float bb = b3[m]; // padded to 16, zeros for m>=8
  for (int mt = 0; mt < 2; ++mt) {
    v16h a0 = frag_a(X, mt * 16 + m, 0,  hi);
    v16h a1 = frag_a(X, mt * 16 + m, 32, hi);
    v8f c = {bb, bb, bb, bb, bb, bb, bb, bb};
    c = __builtin_amdgcn_wmma_f32_16x16x32_f16(false, a0, false, b0, (short)0, c, false, false);
    c = __builtin_amdgcn_wmma_f32_16x16x32_f16(false, a1, false, b1, (short)0, c, false, false);
    if (m < 8) {
      float* orow = sF + (mt * 16 + 8 * hi) * 8 + m;
#pragma unroll
      for (int v = 0; v < 8; ++v)
        orow[v * 8] = c[v];
    }
  }
}

__device__ __forceinline__ void mlp_forward(_Float16* sX, _Float16* sH,
    const _Float16* sW1t, const _Float16* sW2t, const _Float16* sW3t,
    const float* sB1, const float* sB2, const float* sB3, float* sF, int lane) {
  __syncthreads();
  layer64(sX, sW1t, sB1, sH, lane);   // 50(pad64) -> 64, tanh
  __syncthreads();
  layer64(sH, sW2t, sB2, sX, lane);   // 64 -> 64, tanh (reuses sX as h2 buffer)
  __syncthreads();
  layer_out(sX, sW3t, sB3, sF, lane); // 64 -> 8
  __syncthreads();
}

// Grey-box CSTR+flash RHS in scaled coords (port of _fg, native-HW math).
__device__ __forceinline__ void fg_eval(const float x[8], float u0, float u1, float dx[8]) {
  float Hr = x[0] * 0.3f + 0.7f,  CAr = x[1] * 0.2f + 0.5f;
  float CBr = x[2] * 0.2f + 0.5f, Tr  = x[3] * 5.0f + 310.0f;
  float Hb = x[4] * 0.3f + 0.7f,  CAb = x[5] * 0.2f + 0.5f;
  float CBb = x[6] * 0.2f + 0.5f, Tb  = x[7] * 5.0f + 310.0f;
  float F = u0 * 0.1f + 1.0f, D = u1 * 0.05f + 0.5f;
  float invden = __builtin_amdgcn_rcpf(3.5f * CAb + 1.1f * CBb);
  float CAd = 3.5f * CAb * invden, CBd = 1.1f * CBb * invden;
  float Fr = __builtin_amdgcn_sqrtf(Hr), Fb = __builtin_amdgcn_sqrtf(Hb);
  // 20000*exp(-3000/Tr) = 20000*exp2(-3000*log2(e)/Tr)
  float k1 = 20000.0f * __builtin_amdgcn_exp2f(-4328.085122666891f * __builtin_amdgcn_rcpf(Tr));
  float r1 = k1 * CAr;
  float invHr = __builtin_amdgcn_rcpf(Hr), invHb = __builtin_amdgcn_rcpf(Hb);
  float dHr  = F + D - Fr;
  float dCAr = (F * (1.0f - CAr) + D * (CAd - CAr)) * invHr - r1;
  float dCBr = (-F * CBr + D * (CBd - CBr)) * invHr + r1;
  float dTr  = (F * (320.0f - Tr) + D * (310.0f - Tr)) * invHr
             - (200.0f / 15.0f) * invHr + r1 * (10.0f / 15.0f);
  float dHb  = Fr - Fb - D;
  float dCAb = (Fr * (CAr - CAb) + D * (CAb - CAd)) * invHb;
  float dCBb = (Fr * (CBr - CBb) + D * (CBb - CBd)) * invHb;
  float dTb  = Fr * (Tr - Tb) * invHb + (200.0f / 15.0f) * invHb;
  dx[0] = dHr  * (1.0f / 0.3f);  dx[1] = dCAr * (1.0f / 0.2f);
  dx[2] = dCBr * (1.0f / 0.2f);  dx[3] = dTr  * (1.0f / 5.0f);
  dx[4] = dHb  * (1.0f / 0.3f);  dx[5] = dCAb * (1.0f / 0.2f);
  dx[6] = dCBb * (1.0f / 0.2f);  dx[7] = dTb  * (1.0f / 5.0f);
}

__global__ __launch_bounds__(32) void cstr_scan_kernel(
    const float* __restrict__ useq, const float* __restrict__ xGz0,
    const float* __restrict__ W1, const float* __restrict__ b1,
    const float* __restrict__ W2, const float* __restrict__ b2,
    const float* __restrict__ W3, const float* __restrict__ b3,
    float* __restrict__ out, int T) {
  __shared__ __attribute__((aligned(32))) _Float16 sW1t[64 * 64]; // [n][k], k padded 50->64
  __shared__ __attribute__((aligned(32))) _Float16 sW2t[64 * 64]; // [n][k]
  __shared__ __attribute__((aligned(32))) _Float16 sW3t[16 * 64]; // [n][k], n padded 8->16
  __shared__ float sB1[64], sB2[64], sB3[16];
  __shared__ __attribute__((aligned(32))) _Float16 sX[32 * 64];   // stage input / h2
  __shared__ __attribute__((aligned(32))) _Float16 sH[32 * 64];   // h1
  __shared__ float sF[32 * 8];                                    // fnn output

  const int lane = threadIdx.x;
  const int s = blockIdx.x * 32 + lane;

  // Load + transpose + pad weights into LDS (f16), biases (f32).
  for (int i = lane; i < 64 * 64; i += 32) {
    int k = i >> 6, n = i & 63;
    sW1t[n * 64 + k] = (_Float16)(k < 50 ? W1[k * 64 + n] : 0.0f);
    sW2t[n * 64 + k] = (_Float16)W2[k * 64 + n];
  }
  for (int i = lane; i < 16 * 64; i += 32) {
    int n = i >> 6, k = i & 63;
    sW3t[n * 64 + k] = (_Float16)(n < 8 ? W3[k * 8 + n] : 0.0f);
  }
  for (int i = lane; i < 64; i += 32) { sB1[i] = b1[i]; sB2[i] = b2[i]; }
  if (lane < 16) sB3[lane] = (lane < 8) ? b3[lane] : 0.0f;

  // Per-thread state (scaled coords), one sample per lane.
  float xG[8], yp[32], up[8];
  {
    const float* p = xGz0 + (size_t)s * 48;
#pragma unroll
    for (int i = 0; i < 8; ++i)  xG[i] = p[i];
#pragma unroll
    for (int i = 0; i < 32; ++i) yp[i] = p[8 + i];
#pragma unroll
    for (int i = 0; i < 8; ++i)  up[i] = p[40 + i];
  }
  __syncthreads();

#pragma unroll 1
  for (int t = 0; t < T; ++t) {
    const float* uptr = useq + ((size_t)s * T + t) * 2;
    float u0 = uptr[0], u1 = uptr[1];
    if (t + 1 < T) __builtin_prefetch(uptr + 2, 0, 3);

    float acc[8], xcur[8], kk[8];
    _Float16* r = sX + lane * 64;

    // ---- stage 1: input = [xG, ypseq, upseq, u] ----
#pragma unroll
    for (int i = 0; i < 8; ++i)  r[i] = (_Float16)xG[i];
#pragma unroll
    for (int i = 0; i < 32; ++i) r[8 + i] = (_Float16)yp[i];
#pragma unroll
    for (int i = 0; i < 8; ++i)  r[40 + i] = (_Float16)up[i];
    r[48] = (_Float16)u0; r[49] = (_Float16)u1;
#pragma unroll
    for (int i = 50; i < 64; ++i) r[i] = (_Float16)0.0f;
    mlp_forward(sX, sH, sW1t, sW2t, sW3t, sB1, sB2, sB3, sF, lane);
    fg_eval(xG, u0, u1, kk);
#pragma unroll
    for (int i = 0; i < 8; ++i) {
      kk[i] += sF[lane * 8 + i];
      acc[i] = kk[i];
      xcur[i] = xG[i] + 0.5f * RK_DELTA * kk[i];   // x2
    }

    // ---- stage 2: input = [x2, zi, u], zi = midpoint interp of [yp, xG] ++ up ----
#pragma unroll
    for (int i = 0; i < 8; ++i)  r[i] = (_Float16)xcur[i];
#pragma unroll
    for (int i = 0; i < 32; ++i)
      r[8 + i] = (_Float16)(0.5f * (yp[i] + (i < 24 ? yp[i + 8] : xG[i - 24])));
#pragma unroll
    for (int i = 0; i < 8; ++i)  r[40 + i] = (_Float16)up[i];
    r[48] = (_Float16)u0; r[49] = (_Float16)u1;
#pragma unroll
    for (int i = 50; i < 64; ++i) r[i] = (_Float16)0.0f;
    mlp_forward(sX, sH, sW1t, sW2t, sW3t, sB1, sB2, sB3, sF, lane);
    fg_eval(xcur, u0, u1, kk);
#pragma unroll
    for (int i = 0; i < 8; ++i) {
      kk[i] += sF[lane * 8 + i];
      acc[i] += 2.0f * kk[i];
      xcur[i] = xG[i] + 0.5f * RK_DELTA * kk[i];   // x3
    }

    // ---- stage 3: input = [x3, zi, u] (same zi) ----
#pragma unroll
    for (int i = 0; i < 8; ++i)  r[i] = (_Float16)xcur[i];
#pragma unroll
    for (int i = 0; i < 32; ++i)
      r[8 + i] = (_Float16)(0.5f * (yp[i] + (i < 24 ? yp[i + 8] : xG[i - 24])));
#pragma unroll
    for (int i = 0; i < 8; ++i)  r[40 + i] = (_Float16)up[i];
    r[48] = (_Float16)u0; r[49] = (_Float16)u1;
#pragma unroll
    for (int i = 50; i < 64; ++i) r[i] = (_Float16)0.0f;
    mlp_forward(sX, sH, sW1t, sW2t, sW3t, sB1, sB2, sB3, sF, lane);
    fg_eval(xcur, u0, u1, kk);
#pragma unroll
    for (int i = 0; i < 8; ++i) {
      kk[i] += sF[lane * 8 + i];
      acc[i] += 2.0f * kk[i];
      xcur[i] = xG[i] + RK_DELTA * kk[i];          // x4
    }

    // Shift y-history in place: ypShifted = [yp[8:32], xG] (xG still old)
#pragma unroll
    for (int i = 0; i < 24; ++i) yp[i] = yp[i + 8];
#pragma unroll
    for (int i = 0; i < 8; ++i)  yp[24 + i] = xG[i];

    // ---- stage 4: input = [x4, ypShifted, upseq, u] ----
#pragma unroll
    for (int i = 0; i < 8; ++i)  r[i] = (_Float16)xcur[i];
#pragma unroll
    for (int i = 0; i < 32; ++i) r[8 + i] = (_Float16)yp[i];
#pragma unroll
    for (int i = 0; i < 8; ++i)  r[40 + i] = (_Float16)up[i];
    r[48] = (_Float16)u0; r[49] = (_Float16)u1;
#pragma unroll
    for (int i = 50; i < 64; ++i) r[i] = (_Float16)0.0f;
    mlp_forward(sX, sH, sW1t, sW2t, sW3t, sB1, sB2, sB3, sF, lane);
    fg_eval(xcur, u0, u1, kk);
#pragma unroll
    for (int i = 0; i < 8; ++i) {
      kk[i] += sF[lane * 8 + i];
      acc[i] += kk[i];
    }

    // Emit y = xG (pre-update), then RK4 update + u-history shift.
    float* o = out + ((size_t)s * T + t) * 8;
#pragma unroll
    for (int i = 0; i < 8; ++i) o[i] = xG[i];
#pragma unroll
    for (int i = 0; i < 8; ++i) xG[i] += (RK_DELTA / 6.0f) * acc[i];
#pragma unroll
    for (int i = 0; i < 6; ++i) up[i] = up[i + 2];
    up[6] = u0; up[7] = u1;
  }
}

extern "C" void kernel_launch(void* const* d_in, const int* in_sizes, int n_in,
                              void* d_out, int out_size, void* d_ws, size_t ws_size,
                              hipStream_t stream) {
  const float* useq = (const float*)d_in[0];
  const float* xGz0 = (const float*)d_in[1];
  const float* W1   = (const float*)d_in[2];
  const float* b1   = (const float*)d_in[3];
  const float* W2   = (const float*)d_in[4];
  const float* b2   = (const float*)d_in[5];
  const float* W3   = (const float*)d_in[6];
  const float* b3   = (const float*)d_in[7];
  float* out = (float*)d_out;

  const int B = in_sizes[1] / 48;        // 8192
  const int T = in_sizes[0] / (B * 2);   // 256

  dim3 grid(B / 32), block(32);
  cstr_scan_kernel<<<grid, block, 0, stream>>>(useq, xGz0, W1, b1, W2, b2, W3, b3, out, T);
}
